// GraphVAEEncoder_11562051961055
// MI455X (gfx1250) — matrix operand order repaired and checked by reading, not compiled, not run on hardware
//
#include <hip/hip_runtime.h>
#include <hip/hip_bf16.h>

// ---- fixed problem shape (from reference) ----
#define NTOT   65536        // B * N_PER
#define NPER   1024
#define NB     64
#define HID    128
#define LAT    64
#define K1IN   80           // NODE_IN + PE
#define K1PAD  96           // padded to 3 * 32
#define KT1    3            // K-tiles layer 1
#define KT2    4            // K-tiles layer 2 (K=128)
#define NEDGE  1048576

typedef __attribute__((ext_vector_type(16))) __bf16 v16bf;
typedef __attribute__((ext_vector_type(8)))  __bf16 v8bf;
typedef __attribute__((ext_vector_type(8)))  float  v8f;

__device__ __forceinline__ unsigned short f2bf(float f) {
    unsigned u = __float_as_uint(f);
    u += 0x7FFFu + ((u >> 16) & 1u);          // round-to-nearest-even
    return (unsigned short)(u >> 16);
}

// ---------------------------------------------------------------------------
// pack x||pe -> bf16 [NTOT, K1PAD] (zero padded cols 80..95)
__global__ void pack_xin_kernel(const float* __restrict__ x,
                                const float* __restrict__ pe,
                                unsigned short* __restrict__ out, int n) {
    int idx = blockIdx.x * blockDim.x + threadIdx.x;
    if (idx >= n) return;
    int node = idx / K1PAD;
    int c    = idx - node * K1PAD;
    float v = 0.0f;
    if (c < 64)      v = x[(size_t)node * 64 + c];
    else if (c < 80) v = pe[(size_t)node * 16 + (c - 64)];
    out[idx] = f2bf(v);
}

// pack W[K,N] (row major f32) into WMMA B-fragment layout, bf16:
// out[((kt*Ntiles + nt)*32 + lane)*16 + j]  with  k = kt*32 + (lane>>4)*16 + j,
// n = nt*16 + (lane&15).  Zero-pads k >= K.
__global__ void pack_w_kernel(const float* __restrict__ W,
                              unsigned short* __restrict__ out,
                              int K, int N, int total) {
    int idx = blockIdx.x * blockDim.x + threadIdx.x;
    if (idx >= total) return;
    int blk    = idx >> 9;            // 512 elems per (kt,nt) block
    int within = idx & 511;
    int lane   = within >> 4;
    int j      = within & 15;
    int Ntiles = N >> 4;
    int kt = blk / Ntiles;
    int nt = blk - kt * Ntiles;
    int n  = nt * 16 + (lane & 15);
    int k  = kt * 32 + (lane >> 4) * 16 + j;
    float v = (k < K) ? W[(size_t)k * N + n] : 0.0f;
    out[idx] = f2bf(v);
}

// ---------------------------------------------------------------------------
// degree with self loop: deg[i] = 1 + sum_{e: dst==i} ew[e]
__global__ void deg_init_kernel(float* __restrict__ deg, int n) {
    int i = blockIdx.x * blockDim.x + threadIdx.x;
    if (i < n) deg[i] = 1.0f;                 // self-loop weight
}
__global__ void deg_accum_kernel(const long long* __restrict__ dst,
                                 const float* __restrict__ ew,
                                 float* __restrict__ deg, int e) {
    int i = blockIdx.x * blockDim.x + threadIdx.x;
    if (i >= e) return;
    atomicAdd(&deg[(int)dst[i]], ew[i]);
}
__global__ void dis_kernel(const float* __restrict__ deg,
                           float* __restrict__ dis, int n) {
    int i = blockIdx.x * blockDim.x + threadIdx.x;
    if (i >= n) return;
    float d = deg[i];
    dis[i] = (d > 0.0f) ? rsqrtf(fmaxf(d, 1e-12f)) : 0.0f;
}
__global__ void norm_kernel(const long long* __restrict__ src,
                            const long long* __restrict__ dst,
                            const float* __restrict__ ew,
                            const float* __restrict__ dis,
                            float* __restrict__ norm, int e) {
    int i = blockIdx.x * blockDim.x + threadIdx.x;
    if (i >= e) return;
    norm[i] = dis[(int)src[i]] * ew[i] * dis[(int)dst[i]];
}

// ---------------------------------------------------------------------------
// WMMA GEMM:  C[M,128] = A[M, Ktiles*32](bf16) @ Bpk (fragment-packed bf16)
// One wave per 16-row strip; 8 accumulators cover all 128 output columns.
__global__ void __launch_bounds__(128)
gemm_bf16_wmma(const unsigned short* __restrict__ A,
               const unsigned short* __restrict__ Bpk,
               float* __restrict__ C, int M, int Ktiles) {
    const int lane  = threadIdx.x & 31;
    const int wave  = threadIdx.x >> 5;
    const int mtile = blockIdx.x * 4 + wave;
    const int m0    = mtile * 16;
    if (m0 >= M) return;
    const int Kpad = Ktiles * 32;
    const int half = lane >> 4;
    const int mrow = m0 + (lane & 15);

    v8f acc[8];
#pragma unroll
    for (int nt = 0; nt < 8; ++nt)
        acc[nt] = (v8f){0.f, 0.f, 0.f, 0.f, 0.f, 0.f, 0.f, 0.f};

    for (int kt = 0; kt < Ktiles; ++kt) {
        // A fragment: two contiguous 16B chunks per lane
        const unsigned short* pa = A + (size_t)mrow * Kpad + kt * 32 + half * 16;
        if (kt + 1 < Ktiles) __builtin_prefetch(pa + 32, 0, 1);  // global_prefetch_b8
        union { v16bf v; v8bf h[2]; } ua;
        ua.h[0] = *(const v8bf*)(const void*)(pa);
        ua.h[1] = *(const v8bf*)(const void*)(pa + 8);
        const unsigned short* pb = Bpk + (((size_t)kt * 8) * 32 + lane) * 16;
#pragma unroll
        for (int nt = 0; nt < 8; ++nt) {
            v16bf b = *(const v16bf*)(const void*)(pb + (size_t)nt * 32 * 16);
            acc[nt] = __builtin_amdgcn_wmma_f32_16x16x32_bf16(
                false, ua.v, false, b, (short)0, acc[nt], false, false);
        }
    }
    // store D: VGPR r -> row m0 + half*8 + r, col nt*16 + (lane&15)
    const int nbase = lane & 15;
    const int mbase = m0 + half * 8;
#pragma unroll
    for (int nt = 0; nt < 8; ++nt)
#pragma unroll
        for (int r = 0; r < 8; ++r)
            C[(size_t)(mbase + r) * HID + nt * 16 + nbase] = acc[nt][r];
}

// ---------------------------------------------------------------------------
// agg init fused with the self-loop contribution: agg = hW * dis^2
// (replaces a zero pass + saves re-reading hW in finalize)
__global__ void selfloop_init_kernel(const float* __restrict__ hW,
                                     const float* __restrict__ dis,
                                     float* __restrict__ agg, int n) {
    int idx = blockIdx.x * blockDim.x + threadIdx.x;
    if (idx >= n) return;
    int node = idx >> 7;
    float dv = dis[node];
    agg[idx] = hW[idx] * dv * dv;
}

// edge scatter: one wave per edge, 4 features per lane.
// src/dst/norm are wave-uniform -> readfirstlane to scalarize address math.
__global__ void scatter_kernel(const long long* __restrict__ src,
                               const long long* __restrict__ dst,
                               const float* __restrict__ norm,
                               const float* __restrict__ h,
                               float* __restrict__ agg, int e) {
    long long t = (long long)blockIdx.x * blockDim.x + threadIdx.x;
    long long ed = t >> 5;
    if (ed >= e) return;
    int lane = (int)(t & 31);
    int s = __builtin_amdgcn_readfirstlane((int)src[ed]);
    int d = __builtin_amdgcn_readfirstlane((int)dst[ed]);
    float nv = __uint_as_float(
        __builtin_amdgcn_readfirstlane(__float_as_uint(norm[ed])));
    const float4 hv = *(const float4*)(h + (size_t)s * HID + lane * 4);
    float* o = agg + (size_t)d * HID + lane * 4;
    atomicAdd(o + 0, hv.x * nv);
    atomicAdd(o + 1, hv.y * nv);
    atomicAdd(o + 2, hv.z * nv);
    atomicAdd(o + 3, hv.w * nv);
}

// out = relu(agg + bias); mode 0 -> bf16 buffer (next GEMM A operand),
// mode 1 -> f32 written in place into agg (for pooling)
__global__ void finalize_kernel(const float* __restrict__ agg,
                                const float* __restrict__ bias,
                                unsigned short* __restrict__ outBf,
                                float* __restrict__ outF,
                                int mode, int n) {
    int idx = blockIdx.x * blockDim.x + threadIdx.x;
    if (idx >= n) return;
    int f = idx & 127;
    float v = fmaxf(agg[idx] + bias[f], 0.0f);
    if (mode == 0) outBf[idx] = f2bf(v);
    else           outF[idx]  = v;
}

// mean pool: graphs are contiguous blocks of NPER nodes
__global__ void __launch_bounds__(HID)
pool_kernel(const float* __restrict__ h2, float* __restrict__ hg) {
    int g = blockIdx.x;
    int f = threadIdx.x;
    float s = 0.0f;
    const float* p = h2 + (size_t)g * NPER * HID + f;
    for (int n = 0; n < NPER; ++n) s += p[(size_t)n * HID];
    hg[g * HID + f] = s * (1.0f / NPER);
}

__device__ __forceinline__ float hash_normal(unsigned idx) {
    unsigned a = idx * 0x9E3779B9u + 0x85EBCA6Bu;
    a ^= a >> 16; a *= 0x7FEB352Du; a ^= a >> 15; a *= 0x846CA68Bu; a ^= a >> 16;
    unsigned b = (idx ^ 0xDEADBEEFu) * 0x9E3779B9u + 0xC2B2AE35u;
    b ^= b >> 16; b *= 0x7FEB352Du; b ^= b >> 15; b *= 0x846CA68Bu; b ^= b >> 16;
    float u1 = (float)(a >> 8) * (1.0f / 16777216.0f) + (1.0f / 33554432.0f);
    float u2 = (float)(b >> 8) * (1.0f / 16777216.0f);
    return sqrtf(-2.0f * logf(u1)) * cosf(6.28318530718f * u2);
}

// head: mu/logvar projections + reparameterize; out = [z | mu | logvar]
__global__ void head_kernel(const float* __restrict__ hg,
                            const float* __restrict__ Wmu,
                            const float* __restrict__ bmu,
                            const float* __restrict__ Wlv,
                            const float* __restrict__ blv,
                            float* __restrict__ out) {
    int idx = blockIdx.x * blockDim.x + threadIdx.x;   // NB*LAT = 4096
    if (idx >= NB * LAT) return;
    int g = idx >> 6;
    int j = idx & 63;
    const float* hrow = hg + g * HID;
    float mu = bmu[j], lv = blv[j];
    for (int k = 0; k < HID; ++k) {
        float hv = hrow[k];
        mu = fmaf(hv, Wmu[k * LAT + j], mu);
        lv = fmaf(hv, Wlv[k * LAT + j], lv);
    }
    float sd  = expf(0.5f * lv);
    float eps = hash_normal((unsigned)idx);
    out[idx]                 = mu + eps * sd;  // z
    out[NB * LAT + idx]      = mu;
    out[2 * NB * LAT + idx]  = lv;
}

// ---------------------------------------------------------------------------
extern "C" void kernel_launch(void* const* d_in, const int* in_sizes, int n_in,
                              void* d_out, int out_size, void* d_ws, size_t ws_size,
                              hipStream_t stream) {
    (void)in_sizes; (void)n_in; (void)out_size; (void)ws_size;
    const float*     x   = (const float*)d_in[0];
    const long long* ei  = (const long long*)d_in[1];       // int64 [2,E]
    const float*     ew  = (const float*)d_in[2];
    /* d_in[3] = batch (unused; graphs are contiguous) */
    const float*     pe  = (const float*)d_in[4];
    const float*     W1  = (const float*)d_in[5];
    const float*     b1  = (const float*)d_in[6];
    const float*     W2  = (const float*)d_in[7];
    const float*     b2  = (const float*)d_in[8];
    const float*     Wmu = (const float*)d_in[9];
    const float*     bmu = (const float*)d_in[10];
    const float*     Wlv = (const float*)d_in[11];
    const float*     blv = (const float*)d_in[12];
    const long long* src = ei;
    const long long* dst = ei + NEDGE;

    // workspace layout (256B aligned)
    char* ws = (char*)d_ws;
    size_t o = 0;
    auto take = [&](size_t bytes) { char* p = ws + o; o = (o + bytes + 255) & ~(size_t)255; return p; };
    unsigned short* xin_bf = (unsigned short*)take((size_t)NTOT * K1PAD * 2);   // 12.6 MB
    unsigned short* w1pk   = (unsigned short*)take((size_t)K1PAD * HID * 2);
    unsigned short* w2pk   = (unsigned short*)take((size_t)HID * HID * 2);
    float*          deg    = (float*)take((size_t)NTOT * 4);
    float*          dis    = (float*)take((size_t)NTOT * 4);
    float*          norm   = (float*)take((size_t)NEDGE * 4);                   // 4 MB
    float*          hW     = (float*)take((size_t)NTOT * HID * 4);              // 33.5 MB
    float*          agg    = (float*)take((size_t)NTOT * HID * 4);              // 33.5 MB
    unsigned short* h1bf   = (unsigned short*)take((size_t)NTOT * HID * 2);     // 16.8 MB
    float*          hg     = (float*)take((size_t)NB * HID * 4);

    const int T = 256;
    auto cdiv = [](long long a, long long b) { return (unsigned)((a + b - 1) / b); };

    // 1. inputs -> bf16 fragments
    pack_xin_kernel<<<cdiv((long long)NTOT * K1PAD, T), T, 0, stream>>>(x, pe, xin_bf, NTOT * K1PAD);
    pack_w_kernel<<<cdiv(K1PAD * HID, T), T, 0, stream>>>(W1, w1pk, K1IN, HID, K1PAD * HID);
    pack_w_kernel<<<cdiv(HID * HID, T), T, 0, stream>>>(W2, w2pk, HID, HID, HID * HID);

    // 2. degrees / normalization
    deg_init_kernel<<<cdiv(NTOT, T), T, 0, stream>>>(deg, NTOT);
    deg_accum_kernel<<<cdiv(NEDGE, T), T, 0, stream>>>(dst, ew, deg, NEDGE);
    dis_kernel<<<cdiv(NTOT, T), T, 0, stream>>>(deg, dis, NTOT);
    norm_kernel<<<cdiv(NEDGE, T), T, 0, stream>>>(src, dst, ew, dis, norm, NEDGE);

    // 3. layer 1: GEMM (WMMA) -> self-loop init -> scatter -> finalize(relu, ->bf16)
    gemm_bf16_wmma<<<NTOT / 64, 128, 0, stream>>>(xin_bf, w1pk, hW, NTOT, KT1);
    selfloop_init_kernel<<<cdiv((long long)NTOT * HID, T), T, 0, stream>>>(hW, dis, agg, NTOT * HID);
    scatter_kernel<<<cdiv((long long)NEDGE * 32, T), T, 0, stream>>>(src, dst, norm, hW, agg, NEDGE);
    finalize_kernel<<<cdiv((long long)NTOT * HID, T), T, 0, stream>>>(agg, b1, h1bf, agg, 0, NTOT * HID);

    // 4. layer 2: GEMM (WMMA) -> self-loop init -> scatter -> finalize(relu, f32 in place)
    gemm_bf16_wmma<<<NTOT / 64, 128, 0, stream>>>(h1bf, w2pk, hW, NTOT, KT2);
    selfloop_init_kernel<<<cdiv((long long)NTOT * HID, T), T, 0, stream>>>(hW, dis, agg, NTOT * HID);
    scatter_kernel<<<cdiv((long long)NEDGE * 32, T), T, 0, stream>>>(src, dst, norm, hW, agg, NEDGE);
    finalize_kernel<<<cdiv((long long)NTOT * HID, T), T, 0, stream>>>(agg, b2, h1bf, agg, 1, NTOT * HID);

    // 5. pool + VAE head
    pool_kernel<<<NB, HID, 0, stream>>>(agg, hg);
    head_kernel<<<cdiv(NB * LAT, 128), 128, 0, stream>>>(hg, Wmu, bmu, Wlv, blv, (float*)d_out);
}
